// BaseLayer_13967233647354
// MI455X (gfx1250) — matrix operand order repaired and checked by reading, not compiled, not run on hardware
//
#include <hip/hip_runtime.h>

// ---------------------------------------------------------------------------
// BaseLayer (MoE, E=8) fused inference for MI455X / gfx1250.
//   B=8, S=1024, D=1024, F=4096, E=8  ->  T = 8192 tokens.
// Pipeline:
//   k0: zero per-expert counters
//   k1: routing (argmax over centroid dots), sigmoid gate, per-expert
//       LayerNorm, write xln as bf16, bucketize tokens per expert
//   k2: h = relu(xln @ w1[e] + b1[e])           (bf16 WMMA, f32 acc)
//   k3: out = x + alpha * (h @ w2[e] + b2[e])   (bf16 WMMA, f32 acc)
// GEMM: 64x128 block tile, 8 waves (2x4) each owning 32x32 (4 accumulators),
// K step 32, double-buffered LDS (1 barrier / K-step), fragment-contiguous
// layouts so every WMMA operand is 2x ds_load_b128. fp32->bf16 uses the
// native bf16 convert (RNE).
// ---------------------------------------------------------------------------

typedef __attribute__((ext_vector_type(16))) __bf16          v16bf;
typedef __attribute__((ext_vector_type(8)))  float           v8f;
typedef __attribute__((ext_vector_type(8)))  unsigned short  ushort8;

#define DEV static __device__ __forceinline__

constexpr int Bb = 8;
constexpr int Ss = 1024;
constexpr int Dd = 1024;
constexpr int Ff = 4096;
constexpr int Ee = 8;
constexpr int Tt = Bb * Ss;           // 8192 tokens
constexpr float EPS = 1e-5f;

constexpr int LDA = 40;               // LDS row stride in shorts (80 B, 16B-aligned)

union FRAG { ushort8 h[2]; v16bf v; };

DEV unsigned short f2bfu(float f) {   // fp32 -> bf16 via native convert (RNE)
    union { __bf16 b; unsigned short u; } c;
    c.b = (__bf16)f;
    return c.u;
}

DEV v8f wmma_bf16(const FRAG& a, const FRAG& b, v8f c) {
    return __builtin_amdgcn_wmma_f32_16x16x32_bf16(
               false, a.v, false, b.v, (short)0, c, false, false);
}

// ---------------------------------------------------------------------------
__global__ __launch_bounds__(64)
void init_kernel(int* __restrict__ cnt) {
    if (threadIdx.x < 16) cnt[threadIdx.x] = 0;
}

// ---------------------------------------------------------------------------
// One wave32 per token: routing + gate + LayerNorm, bf16 write of xln.
// ---------------------------------------------------------------------------
__global__ __launch_bounds__(256)
void route_ln_kernel(const float* __restrict__ x,
                     const float* __restrict__ cent,
                     const float* __restrict__ ln_g,
                     const float* __restrict__ ln_b,
                     unsigned short* __restrict__ xln,
                     float* __restrict__ alpha,
                     int* __restrict__ cnt,
                     int* __restrict__ toklist)
{
    const int lane = threadIdx.x & 31;
    const int wave = threadIdx.x >> 5;
    const int t    = blockIdx.x * 8 + wave;

    const float* xr = x + (size_t)t * Dd;

    float xv[32];
    #pragma unroll
    for (int j = 0; j < 32; ++j) xv[j] = xr[lane + 32 * j];

    float s = 0.f, ss = 0.f;
    #pragma unroll
    for (int j = 0; j < 32; ++j) { s += xv[j]; ss += xv[j] * xv[j]; }

    float dot[Ee];
    #pragma unroll
    for (int e = 0; e < Ee; ++e) {
        const float* c = cent + (size_t)e * Dd;
        float d = 0.f;
        #pragma unroll
        for (int j = 0; j < 32; ++j) d += xv[j] * c[lane + 32 * j];
        dot[e] = d;
    }

    #pragma unroll
    for (int off = 16; off > 0; off >>= 1) {
        s  += __shfl_xor(s,  off, 32);
        ss += __shfl_xor(ss, off, 32);
        #pragma unroll
        for (int e = 0; e < Ee; ++e) dot[e] += __shfl_xor(dot[e], off, 32);
    }

    int   eid  = 0;
    float best = dot[0];
    #pragma unroll
    for (int e = 1; e < Ee; ++e)
        if (dot[e] > best) { best = dot[e]; eid = e; }

    const float mu   = s * (1.f / Dd);
    const float var  = ss * (1.f / Dd) - mu * mu;
    const float rstd = rsqrtf(var + EPS);

    const float* ge = ln_g + (size_t)eid * Dd;
    const float* be = ln_b + (size_t)eid * Dd;
    unsigned short* o = xln + (size_t)t * Dd;
    #pragma unroll
    for (int j = 0; j < 32; ++j) {
        const int i = lane + 32 * j;
        o[i] = f2bfu((xv[j] - mu) * rstd * ge[i] + be[i]);
    }

    if (lane == 0) {
        alpha[t] = 1.f / (1.f + __expf(-best));
        const int pos = atomicAdd(&cnt[eid], 1);
        toklist[eid * Tt + pos] = t;
    }
}

// ---------------------------------------------------------------------------
// Expert GEMM, 64x128 block tile, 8 waves (2x4), wave tile 32x32 (4 accs).
// Double-buffered LDS: one barrier per K-step; global loads for tile k+1
// issued before the WMMAs of tile k.
//   FIRST=true : A=xln[T,K=D],  W=w1[e][K=D,N=F],  relu+bias -> h (bf16)
//   FIRST=false: A=h[T,K=F],    W=w2[e][K=F,N=D],  out = x + alpha*(acc+bias)
// ---------------------------------------------------------------------------
template<int KDIM, int NDIM, bool FIRST>
__global__ __launch_bounds__(256)
void ffn_kernel(const unsigned short* __restrict__ Amat,
                const float* __restrict__ W,
                const float* __restrict__ bias,
                const int* __restrict__ cnt,
                const int* __restrict__ toklist,
                unsigned short* __restrict__ hout,
                const float* __restrict__ x,
                const float* __restrict__ alpha,
                float* __restrict__ out)
{
    const int e     = blockIdx.z;
    const int count = cnt[e];
    const int rowB  = blockIdx.x * 64;
    if (rowB >= count) return;
    const int nB    = blockIdx.y * 128;

    __shared__ __attribute__((aligned(16))) unsigned short As[2][64][LDA];   // [buf][row][k]
    __shared__ __attribute__((aligned(16))) unsigned short Bt[2][128][LDA];  // [buf][n][k]
    __shared__ int tk[64];

    const int tid  = threadIdx.x;
    const int lane = tid & 31;
    const int wave = tid >> 5;
    const int mw   = wave >> 2;     // 0..1
    const int nw   = wave & 3;      // 0..3

    if (tid < 64) {
        const int slot = rowB + tid;
        tk[tid] = (slot < count) ? toklist[e * Tt + slot] : -1;
    }
    __syncthreads();

    // staging assignments
    const int arow = tid >> 2;            // 0..63
    const int akg  = (tid & 3) * 8;       // k-group within 32
    const int bcol = tid & 127;           // n within tile
    const int bk0  = (tid >> 7) * 16;     // k-group base (0 or 16)
    const int at   = tk[arow];

    // fragment coordinates (constant across K loop)
    const int r0   = mw * 32 + (lane & 15);
    const int r1   = r0 + 16;
    const int ka0  = (lane < 16) ? 0 : 8;
    const int ka1  = (lane < 16) ? 16 : 24;
    const int col0 = nw * 32 + (lane & 15);
    const int col1 = col0 + 16;
    const int kb   = (lane < 16) ? 0 : 8;

    v8f acc00 = {0.f,0.f,0.f,0.f,0.f,0.f,0.f,0.f};
    v8f acc01 = acc00, acc10 = acc00, acc11 = acc00;

    const ushort8 zero8 = {0,0,0,0,0,0,0,0};
    const unsigned short* aSrc =
        (at >= 0) ? (Amat + (size_t)at * KDIM + akg) : nullptr;
    const float* bSrc =
        W + (size_t)e * KDIM * NDIM + (size_t)bk0 * NDIM + nB + bcol;

    constexpr int NK = KDIM / 32;

    // ---- prologue: stage tile 0 ----
    {
        ushort8 av = zero8;
        if (at >= 0) av = *(const ushort8*)aSrc;
        *(ushort8*)&As[0][arow][akg] = av;
        ushort8 t0v, t1v;
        #pragma unroll
        for (int q = 0; q < 8; ++q) {
            t0v[q] = f2bfu(bSrc[(size_t)q * NDIM]);
            t1v[q] = f2bfu(bSrc[(size_t)(q + 8) * NDIM]);
        }
        *(ushort8*)&Bt[0][bcol][bk0]     = t0v;
        *(ushort8*)&Bt[0][bcol][bk0 + 8] = t1v;
    }
    __syncthreads();

    for (int kk = 0; kk < NK; ++kk) {
        const int cur = kk & 1;
        const int nxt = cur ^ 1;

        // ---- issue global loads for tile kk+1 (hide under WMMAs) ----
        ushort8 av = zero8;
        float   bv[16];
        if (kk + 1 < NK) {
            if (at >= 0) av = *(const ushort8*)(aSrc + (kk + 1) * 32);
            const float* bs = bSrc + (size_t)(kk + 1) * 32 * NDIM;
            #pragma unroll
            for (int q = 0; q < 16; ++q) bv[q] = bs[(size_t)q * NDIM];
            __builtin_prefetch(bs + 32 * (size_t)NDIM, 0, 1);   // global_prefetch
        }

        // ---- fragments from current buffer: 2x ds_load_b128 each ----
        FRAG fa0, fa1, fb0, fb1;
        fa0.h[0] = *(const ushort8*)&As[cur][r0][ka0];
        fa0.h[1] = *(const ushort8*)&As[cur][r0][ka1];
        fa1.h[0] = *(const ushort8*)&As[cur][r1][ka0];
        fa1.h[1] = *(const ushort8*)&As[cur][r1][ka1];
        fb0.h[0] = *(const ushort8*)&Bt[cur][col0][kb];
        fb0.h[1] = *(const ushort8*)&Bt[cur][col0][kb + 16];
        fb1.h[0] = *(const ushort8*)&Bt[cur][col1][kb];
        fb1.h[1] = *(const ushort8*)&Bt[cur][col1][kb + 16];

        acc00 = wmma_bf16(fa0, fb0, acc00);
        acc01 = wmma_bf16(fa0, fb1, acc01);
        acc10 = wmma_bf16(fa1, fb0, acc10);
        acc11 = wmma_bf16(fa1, fb1, acc11);

        // ---- store tile kk+1 into the other buffer ----
        if (kk + 1 < NK) {
            *(ushort8*)&As[nxt][arow][akg] = av;
            ushort8 t0v, t1v;
            #pragma unroll
            for (int q = 0; q < 8; ++q) {
                t0v[q] = f2bfu(bv[q]);
                t1v[q] = f2bfu(bv[q + 8]);
            }
            *(ushort8*)&Bt[nxt][bcol][bk0]     = t0v;
            *(ushort8*)&Bt[nxt][bcol][bk0 + 8] = t1v;
        }
        __syncthreads();
    }

    // ---- epilogue ----
    const int N  = lane & 15;
    const int n0 = nB + nw * 32 + N;
    const int n1 = n0 + 16;
    const float bv0 = bias[(size_t)e * NDIM + n0];
    const float bv1 = bias[(size_t)e * NDIM + n1];
    #pragma unroll
    for (int i = 0; i < 8; ++i) {
        const int M   = i + ((lane < 16) ? 0 : 8);
        const int sl0 = mw * 32 + M;
        const int sl1 = sl0 + 16;
        const int t0  = tk[sl0];
        const int t1  = tk[sl1];
        if (FIRST) {
            if (t0 >= 0) {
                hout[(size_t)t0 * Ff + n0] = f2bfu(fmaxf(acc00[i] + bv0, 0.f));
                hout[(size_t)t0 * Ff + n1] = f2bfu(fmaxf(acc01[i] + bv1, 0.f));
            }
            if (t1 >= 0) {
                hout[(size_t)t1 * Ff + n0] = f2bfu(fmaxf(acc10[i] + bv0, 0.f));
                hout[(size_t)t1 * Ff + n1] = f2bfu(fmaxf(acc11[i] + bv1, 0.f));
            }
        } else {
            if (t0 >= 0) {
                const float a0 = alpha[t0];
                out[(size_t)t0 * Dd + n0] =
                    x[(size_t)t0 * Dd + n0] + a0 * (acc00[i] + bv0);
                out[(size_t)t0 * Dd + n1] =
                    x[(size_t)t0 * Dd + n1] + a0 * (acc01[i] + bv1);
            }
            if (t1 >= 0) {
                const float a1 = alpha[t1];
                out[(size_t)t1 * Dd + n0] =
                    x[(size_t)t1 * Dd + n0] + a1 * (acc10[i] + bv0);
                out[(size_t)t1 * Dd + n1] =
                    x[(size_t)t1 * Dd + n1] + a1 * (acc11[i] + bv1);
            }
        }
    }
}

// ---------------------------------------------------------------------------
extern "C" void kernel_launch(void* const* d_in, const int* in_sizes, int n_in,
                              void* d_out, int out_size, void* d_ws, size_t ws_size,
                              hipStream_t stream)
{
    const float* x    = (const float*)d_in[0];   // [B,S,D]
    const float* cent = (const float*)d_in[1];   // [E,D]
    const float* ln_g = (const float*)d_in[2];   // [E,D]
    const float* ln_b = (const float*)d_in[3];   // [E,D]
    const float* w1   = (const float*)d_in[4];   // [E,D,F]
    const float* b1   = (const float*)d_in[5];   // [E,F]
    const float* w2   = (const float*)d_in[6];   // [E,F,D]
    const float* b2   = (const float*)d_in[7];   // [E,D]
    float* out        = (float*)d_out;           // [B,S,D]

    char* ws = (char*)d_ws;
    size_t off = 0;
    auto alloc = [&](size_t bytes) -> void* {
        off = (off + 255) & ~(size_t)255;
        void* p = ws + off;
        off += bytes;
        return p;
    };
    unsigned short* xln  = (unsigned short*)alloc((size_t)Tt * Dd * sizeof(unsigned short));
    unsigned short* hbuf = (unsigned short*)alloc((size_t)Tt * Ff * sizeof(unsigned short));
    float*          alp  = (float*)alloc((size_t)Tt * sizeof(float));
    int*            cnt  = (int*)alloc(16 * sizeof(int));
    int*            tok  = (int*)alloc((size_t)Ee * Tt * sizeof(int));
    (void)ws_size; (void)in_sizes; (void)n_in; (void)out_size;

    init_kernel<<<1, 64, 0, stream>>>(cnt);
    route_ln_kernel<<<Tt / 8, 256, 0, stream>>>(x, cent, ln_g, ln_b, xln, alp, cnt, tok);
    ffn_kernel<Dd, Ff, true><<<dim3(Tt / 64, Ff / 128, Ee), 256, 0, stream>>>(
        xln, w1, b1, cnt, tok, hbuf, nullptr, nullptr, nullptr);
    ffn_kernel<Ff, Dd, false><<<dim3(Tt / 64, Dd / 128, Ee), 256, 0, stream>>>(
        hbuf, w2, b2, cnt, tok, nullptr, x, alp, out);
}
